// NeuroplasticityModule_27882927685863
// MI455X (gfx1250) — matrix operand order repaired and checked by reading, not compiled, not run on hardware
//
#include <hip/hip_runtime.h>
#include <hip/hip_bf16.h>
#include <math.h>

// CDNA5 (gfx1250) wave32 WMMA types
typedef __attribute__((ext_vector_type(16))) __bf16 v16bf;
typedef __attribute__((ext_vector_type(8)))  __bf16 v8bf;
typedef __attribute__((ext_vector_type(8)))  float  v8f;

#define BDIM 16384
#define HDIM 2048

// ---------- prep: x (f32) -> bf16 ----------
__global__ void cvt_x_kernel(const float* __restrict__ x, __bf16* __restrict__ xb) {
    size_t i = ((size_t)blockIdx.x * blockDim.x + threadIdx.x) * 8;
#pragma unroll
    for (int j = 0; j < 8; ++j) xb[i + j] = (__bf16)x[i + j];
}

// ---------- prep: Bw = bf16(W), Bm = bf16(mask * pm[k]) ----------
// Both stay row-major [n][k]; B-fragment loads are then contiguous along k.
__global__ void prep_b_kernel(const float* __restrict__ W, const float* __restrict__ mask,
                              const float* __restrict__ pm,
                              __bf16* __restrict__ Bw, __bf16* __restrict__ Bm) {
    size_t i = ((size_t)blockIdx.x * blockDim.x + threadIdx.x) * 8;
    int k = (int)(i & (HDIM - 1));
#pragma unroll
    for (int j = 0; j < 8; ++j) {
        Bw[i + j] = (__bf16)W[i + j];
        Bm[i + j] = (__bf16)(mask[i + j] * pm[k + j]);
    }
}

// ---------- fused dual-GEMM + epilogue ----------
// Block = 128 threads = 4 waves. Each wave computes a 64(M)x32(N) output tile
// with two accumulator sets (W-side and mask-side): 16 WMMAs per k=32 step
// against 16 b128 loads -> 1.0 load per WMMA.
__global__ void __launch_bounds__(128)
fused_neuro_kernel(const __bf16* __restrict__ xb,
                   const __bf16* __restrict__ Bw,
                   const __bf16* __restrict__ Bm,
                   const float*  __restrict__ x,
                   const float*  __restrict__ bias,
                   const float*  __restrict__ pm,
                   float* __restrict__ out) {
    const int lane = threadIdx.x & 31;
    const int wave = threadIdx.x >> 5;
    const int row0 = blockIdx.x * 256 + wave * 64;   // M base of this wave (64 rows)
    const int col0 = blockIdx.y * 32;                // N base (32 cols)
    const int ml     = lane & 15;
    const int khalfA = (lane >> 4) << 4;             // A frag: lane-half K offset (0 / 16)
    const int kB     = (lane >> 4) << 4;             // B frag: lane-half K offset (0 / 16)

    // A rows: four 16-row subtiles; B columns: two 16-col subtiles per side.
    const __bf16* aptr[4];
    aptr[0] = xb + (size_t)(row0 + ml) * HDIM;
    aptr[1] = aptr[0] + (size_t)16 * HDIM;
    aptr[2] = aptr[0] + (size_t)32 * HDIM;
    aptr[3] = aptr[0] + (size_t)48 * HDIM;
    const __bf16* bw0 = Bw + (size_t)(col0 + ml) * HDIM + kB;
    const __bf16* bw1 = bw0 + (size_t)16 * HDIM;
    const __bf16* bm0 = Bm + (size_t)(col0 + ml) * HDIM + kB;
    const __bf16* bm1 = bm0 + (size_t)16 * HDIM;

    v8f accW[4][2] = {};
    v8f accM[4][2] = {};

    for (int k = 0; k < HDIM; k += 32) {
        // Unconditional speculative prefetch of next k-tile (dropped if OOB).
        __builtin_prefetch(aptr[0] + k + 32 + khalfA, 0, 0);
        __builtin_prefetch(aptr[2] + k + 32 + khalfA, 0, 0);
        __builtin_prefetch(bw0 + k + 32, 0, 0);
        __builtin_prefetch(bm0 + k + 32, 0, 0);

        // A fragments: contiguous K order matching B — two 8-element
        // (16-byte) runs per lane: K = khalf..khalf+7 and khalf+8..+15.
        v16bf a[4], bwf[2], bmf[2];
#pragma unroll
        for (int s = 0; s < 4; ++s) {
            union { v16bf v; v8bf h[2]; } u;
            u.h[0] = *(const v8bf*)(aptr[s] + k + khalfA);
            u.h[1] = *(const v8bf*)(aptr[s] + k + khalfA + 8);
            a[s] = u.v;
        }
        // B fragments: 16 contiguous bf16 (32B) per lane, lane-half selects K half.
        bwf[0] = *(const v16bf*)(bw0 + k);
        bwf[1] = *(const v16bf*)(bw1 + k);
        bmf[0] = *(const v16bf*)(bm0 + k);
        bmf[1] = *(const v16bf*)(bm1 + k);

#pragma unroll
        for (int s = 0; s < 4; ++s) {
#pragma unroll
            for (int t = 0; t < 2; ++t) {
                accW[s][t] = __builtin_amdgcn_wmma_f32_16x16x32_bf16(
                    false, a[s], false, bwf[t], (short)0, accW[s][t], false, false);
                accM[s][t] = __builtin_amdgcn_wmma_f32_16x16x32_bf16(
                    false, a[s], false, bmf[t], (short)0, accM[s][t], false, false);
            }
        }
    }

    // Epilogue: C layout — VGPR r: lane<16 -> M=r, lane>=16 -> M=r+8; N = lane&15.
    const int cm = (lane >> 4) << 3;
    const int cn = lane & 15;
    const float scale = 0.5f / (float)HDIM;   // DMN_FACTOR / H
#pragma unroll
    for (int s = 0; s < 4; ++s) {
#pragma unroll
        for (int t = 0; t < 2; ++t) {
#pragma unroll
            for (int r = 0; r < 8; ++r) {
                const int gi = row0 + s * 16 + cm + r;
                const int gj = col0 + t * 16 + cn;
                const float xv = x[(size_t)gi * HDIM + gj];
                const float xs = xv * pm[gj];
                const float z  = accW[s][t][r] + bias[gj];
                const float sig = 1.0f / (1.0f + __expf(-z));
                const float fi  = xs * accM[s][t][r] * scale;
                out[(size_t)gi * HDIM + gj] = xs + sig * fi;
            }
        }
    }
}

extern "C" void kernel_launch(void* const* d_in, const int* in_sizes, int n_in,
                              void* d_out, int out_size, void* d_ws, size_t ws_size,
                              hipStream_t stream) {
    const float* x    = (const float*)d_in[0];   // [B,H]
    const float* W    = (const float*)d_in[1];   // [H,H]
    const float* bias = (const float*)d_in[2];   // [H]
    const float* pm   = (const float*)d_in[3];   // [H]
    const float* mask = (const float*)d_in[4];   // [H,H]
    float* out = (float*)d_out;

    // workspace: x_bf16 (67MB) | Bw (8.4MB) | Bm (8.4MB)  -> whole set fits in 192MB L2
    __bf16* xb = (__bf16*)d_ws;
    __bf16* Bw = xb + (size_t)BDIM * HDIM;
    __bf16* Bm = Bw + (size_t)HDIM * HDIM;

    cvt_x_kernel<<<(BDIM * HDIM / 8) / 256, 256, 0, stream>>>(x, xb);
    prep_b_kernel<<<(HDIM * HDIM / 8) / 256, 256, 0, stream>>>(W, mask, pm, Bw, Bm);

    dim3 grid(BDIM / 256, HDIM / 32);   // 64 x 64 blocks
    fused_neuro_kernel<<<grid, 128, 0, stream>>>(xb, Bw, Bm, x, bias, pm, out);
}
